// MyMultiHeadAttention_4148938408938
// MI455X (gfx1250) — matrix operand order repaired and checked by reading, not compiled, not run on hardware
//
#include <hip/hip_runtime.h>

typedef __attribute__((ext_vector_type(16))) _Float16 v16h;
typedef __attribute__((ext_vector_type(8)))  _Float16 v8h;
typedef __attribute__((ext_vector_type(4)))  _Float16 v4h;
typedef __attribute__((ext_vector_type(8)))  float    v8f;
typedef __attribute__((ext_vector_type(4)))  int      v4i;

constexpr int Bb = 2, Ss = 2048, Ee = 1024, Hh = 16, HDd = 64;
constexpr int Mtot = Bb * Ss;  // 4096

#if __has_builtin(__builtin_amdgcn_global_load_async_to_lds_b128) && \
    __has_builtin(__builtin_amdgcn_s_wait_asynccnt)
#define HAVE_ASYNC_LDS 1
#else
#define HAVE_ASYNC_LDS 0
#endif

// 16-byte global -> LDS copy: async (GLOBAL_LOAD_ASYNC_TO_LDS_B128, ASYNCcnt)
// when available, else synchronous load+ds_store fallback.
__device__ __forceinline__ void copy16_g2l(const _Float16* g, _Float16* l) {
#if HAVE_ASYNC_LDS
  __builtin_amdgcn_global_load_async_to_lds_b128(
      (__attribute__((address_space(1))) v4i*)(g),
      (__attribute__((address_space(3))) v4i*)(l), 0, 0);
#else
  *(v8h*)l = *(const v8h*)g;
#endif
}

__device__ __forceinline__ void wait_async_lds() {
#if HAVE_ASYNC_LDS
  __builtin_amdgcn_s_wait_asynccnt(0);
#endif
}

// ---------------------------------------------------------------------------
// WMMA fragment loaders (CDNA5 ISA 7.12.2 VGPR layouts, wave32).
// Work for both global and LDS source pointers (ds_load_b128 from LDS).
// ---------------------------------------------------------------------------
__device__ __forceinline__ v16h load_a_frag(const _Float16* __restrict__ p, int ldk,
                                            int row0, int k0, int lane) {
  const int row = row0 + (lane & 15);
  const int kb  = k0 + ((lane >> 4) << 3);
  const _Float16* base = p + (size_t)row * ldk + kb;
  union { v16h v; v8h h[2]; } u;
  u.h[0] = *(const v8h*)(base);
  u.h[1] = *(const v8h*)(base + 16);
  return u.v;
}

// B 32x16 (KxN) fragment from a row-major [N][K] ("transposed") source.
__device__ __forceinline__ v16h load_bT_frag(const _Float16* __restrict__ p, int ldk,
                                             int n0, int k0, int lane) {
  const int col = n0 + (lane & 15);
  const int kb  = k0 + ((lane >> 4) << 3);
  const _Float16* base = p + (size_t)col * ldk + kb;
  union { v16h v; v8h h[2]; } u;
  u.h[0] = *(const v8h*)(base);
  u.h[1] = *(const v8h*)(base + 16);
  return u.v;
}

__device__ __forceinline__ v8f wmma_f16(v16h a, v16h b, v8f c) {
  return __builtin_amdgcn_wmma_f32_16x16x32_f16(false, a, false, b, (short)0, c,
                                                false, false);
}

// ---------------------------------------------------------------------------
// f32 -> f16 cast (4-wide)
// ---------------------------------------------------------------------------
__global__ void cast_f32_to_f16(const float* __restrict__ in,
                                _Float16* __restrict__ out, int n) {
  const int i = (blockIdx.x * blockDim.x + threadIdx.x) * 4;
  if (i < n) {
    const float4 f = *(const float4*)(in + i);
    v4h h;
    h.x = (_Float16)f.x; h.y = (_Float16)f.y;
    h.z = (_Float16)f.z; h.w = (_Float16)f.w;
    *(v4h*)(out + i) = h;
  }
}

// ---------------------------------------------------------------------------
// GEMM: Y[m,n] = sum_k A[m,k] * W[n,k] + bias[n]   (x @ W^T + b)
// 256-thread block computes 128x128; async-LDS double-buffered K-steps of 32.
// Wave (8 total, 4x2 grid) computes 32(M) x 64(N) = 8 WMMA accumulators.
// mode 0: f16 out [B,H,S,HD] (Q/K);  mode 1: f16 out [B,H,HD,S] (V^T);
// mode 2: f32 out row-major [M,N] (final projection -> d_out)
// ---------------------------------------------------------------------------
__global__ void __launch_bounds__(256)
gemm_wmma_kernel(const _Float16* __restrict__ A, const _Float16* __restrict__ W,
                 const float* __restrict__ bias, _Float16* __restrict__ outh,
                 float* __restrict__ outf, int M, int N, int K, int mode,
                 float oscale) {
  __shared__ _Float16 Atile[2][128 * 32];
  __shared__ _Float16 Btile[2][128 * 32];

  const int tid  = threadIdx.x;
  const int lane = tid & 31;
  const int wave = tid >> 5;
  const int tilesN = N >> 7;
  const int m0 = (blockIdx.x / tilesN) << 7;
  const int n0 = (blockIdx.x % tilesN) << 7;
  const int m_off = (wave & 3) << 5;  // 0,32,64,96
  const int n_off = (wave >> 2) << 6; // 0,64

  const int seg = tid & 3;   // 16B segment within a 64B (32-half) row
  const int r0  = tid >> 2;  // 0..63

  auto load_tiles = [&](int buf, int k0) {
#pragma unroll
    for (int pass = 0; pass < 2; ++pass) {
      const int row = r0 + 64 * pass;
      copy16_g2l(A + (size_t)(m0 + row) * K + k0 + seg * 8,
                 &Atile[buf][row * 32 + seg * 8]);
      copy16_g2l(W + (size_t)(n0 + row) * K + k0 + seg * 8,
                 &Btile[buf][row * 32 + seg * 8]);
    }
  };

  v8f c[2][4];
#pragma unroll
  for (int nf = 0; nf < 4; ++nf) {
    const float bv = bias[n0 + n_off + 16 * nf + (lane & 15)];
#pragma unroll
    for (int r = 0; r < 8; ++r) { c[0][nf][r] = bv; c[1][nf][r] = bv; }
  }

  const int nIter = K >> 5;
  load_tiles(0, 0);
  for (int it = 0; it < nIter; ++it) {
    const int cur = it & 1;
    wait_async_lds();     // my buffer-cur copies have landed
    __syncthreads();      // everyone's copies landed; prev reads done
    if (it + 1 < nIter) load_tiles(cur ^ 1, (it + 1) << 5);  // overlap

    const _Float16* At = &Atile[cur][0];
    const _Float16* Bt = &Btile[cur][0];
    const v16h a0 = load_a_frag(At, 32, m_off,      0, lane);
    const v16h a1 = load_a_frag(At, 32, m_off + 16, 0, lane);
#pragma unroll
    for (int nf = 0; nf < 4; ++nf) {
      const v16h b = load_bT_frag(Bt, 32, n_off + 16 * nf, 0, lane);
      c[0][nf] = wmma_f16(a0, b, c[0][nf]);
      c[1][nf] = wmma_f16(a1, b, c[1][nf]);
    }
  }

#pragma unroll
  for (int mf = 0; mf < 2; ++mf) {
#pragma unroll
    for (int nf = 0; nf < 4; ++nf) {
      const int n = n0 + n_off + 16 * nf + (lane & 15);
#pragma unroll
      for (int r = 0; r < 8; ++r) {
        const int m = m0 + m_off + 16 * mf + ((lane & 16) ? (8 + r) : r);
        const float val = c[mf][nf][r] * oscale;
        if (mode == 2) {
          outf[(size_t)m * N + n] = val;
        } else {
          const int bi = m >> 11, s = m & (Ss - 1);
          const int h = n >> 6, d = n & (HDd - 1);
          const size_t idx =
              (mode == 0) ? ((((size_t)bi * Hh + h) * Ss + s) * HDd + d)
                          : ((((size_t)bi * Hh + h) * HDd + d) * Ss + s);
          outh[idx] = (_Float16)val;
        }
      }
    }
  }
}

// ---------------------------------------------------------------------------
// Flash attention. 256-thread block = 8 waves, all on the SAME (b,h); K/V
// tiles (32 keys) staged once per block into LDS via async copies and shared
// by all 8 waves (8x vmem reduction). Scores computed TRANSPOSED (S^T=K*Q^T)
// so score-accumulator C/D layout == P A-frag layout (no transpose). Q was
// pre-scaled by log2(e)/sqrt(HD) so softmax uses exp2 (v_exp_f32).
// ---------------------------------------------------------------------------
__global__ void __launch_bounds__(256)
flash_attn_kernel(const _Float16* __restrict__ Qh, const _Float16* __restrict__ Kh,
                  const _Float16* __restrict__ Vt, _Float16* __restrict__ outh) {
  __shared__ _Float16 Ktile[2][32 * 64];  // [key][d]
  __shared__ _Float16 Vtile[2][64 * 32];  // [d][key]

  const int tid  = threadIdx.x;
  const int lane = tid & 31;
  const int wave = tid >> 5;
  const int bh = blockIdx.x >> 4;                      // 16 blocks per (b,h)
  const int q0 = (((blockIdx.x & 15) << 3) + wave) << 4;
  const _Float16* Qp = Qh + (size_t)bh * Ss * HDd;
  const _Float16* Kp = Kh + (size_t)bh * Ss * HDd;
  const _Float16* Vp = Vt + (size_t)bh * HDd * Ss;

  // Q^T B-frags (B[d][q] = Q[q][d]) preloaded once per wave.
  const v16h bq0 = load_bT_frag(Qp, HDd, q0, 0, lane);
  const v16h bq1 = load_bT_frag(Qp, HDd, q0, 32, lane);

  const int krow = tid >> 3, kseg = tid & 7;  // 32 rows x 64 halves
  const int vrow = tid >> 2, vseg = tid & 3;  // 64 rows x 32 halves
  auto stage = [&](int buf, int kb) {
    copy16_g2l(Kp + (size_t)(kb + krow) * HDd + kseg * 8,
               &Ktile[buf][krow * 64 + kseg * 8]);
    copy16_g2l(Vp + (size_t)vrow * Ss + kb + vseg * 8,
               &Vtile[buf][vrow * 32 + vseg * 8]);
  };

  v8f o[4];
#pragma unroll
  for (int nf = 0; nf < 4; ++nf)
#pragma unroll
    for (int r = 0; r < 8; ++r) o[nf][r] = 0.f;
  float m_i = -1e30f, l_i = 0.f;

  const int nIter = Ss >> 5;
  stage(0, 0);
  for (int it = 0; it < nIter; ++it) {
    const int cur = it & 1;
    wait_async_lds();
    __syncthreads();
    if (it + 1 < nIter) stage(cur ^ 1, (it + 1) << 5);

    const _Float16* Kb = &Ktile[cur][0];
    const _Float16* Vb = &Vtile[cur][0];

    // S^T tiles: s0 = keys 0..15 of step, s1 = keys 16..31 (cols = queries)
    v8f s0 = {}, s1 = {};
    {
      v16h ka = load_a_frag(Kb, 64, 0, 0, lane);
      s0 = wmma_f16(ka, bq0, s0);
      ka = load_a_frag(Kb, 64, 0, 32, lane);
      s0 = wmma_f16(ka, bq1, s0);
      ka = load_a_frag(Kb, 64, 16, 0, lane);
      s1 = wmma_f16(ka, bq0, s1);
      ka = load_a_frag(Kb, 64, 16, 32, lane);
      s1 = wmma_f16(ka, bq1, s1);
    }
    // Per-query max: 16 in-lane values + pair-lane (xor 16) combine.
    float mx = s0[0];
#pragma unroll
    for (int r = 1; r < 8; ++r) mx = fmaxf(mx, s0[r]);
#pragma unroll
    for (int r = 0; r < 8; ++r) mx = fmaxf(mx, s1[r]);
    mx = fmaxf(mx, __shfl_xor(mx, 16, 32));
    const float m_new = fmaxf(m_i, mx);
    const float alpha = exp2f(m_i - m_new);

    float p[16];
#pragma unroll
    for (int r = 0; r < 8; ++r) {
      p[r]     = exp2f(s0[r] - m_new);
      p[8 + r] = exp2f(s1[r] - m_new);
    }
    float rs = 0.f;
#pragma unroll
    for (int i = 0; i < 16; ++i) rs += p[i];
    rs += __shfl_xor(rs, 16, 32);
    l_i = l_i * alpha + rs;
    m_i = m_new;

    // S^T C-layout == P A-frag layout: pack directly.
    v16h pa;
#pragma unroll
    for (int i = 0; i < 16; ++i) pa[i] = (_Float16)p[i];

#pragma unroll
    for (int r = 0; r < 8; ++r) {
      const float ar = __shfl(alpha, (lane & 16) ? (8 + r) : r, 32);
#pragma unroll
      for (int nf = 0; nf < 4; ++nf) o[nf][r] *= ar;
    }
#pragma unroll
    for (int nf = 0; nf < 4; ++nf) {
      const v16h vb = load_bT_frag(Vb, 32, nf * 16, 0, lane);
      o[nf] = wmma_f16(pa, vb, o[nf]);
    }
  }

  const float linv = 1.f / l_i;
  const int b = bh >> 4, h = bh & 15;
#pragma unroll
  for (int r = 0; r < 8; ++r) {
    const float lr = __shfl(linv, (lane & 16) ? (8 + r) : r, 32);
    const int q = q0 + ((lane & 16) ? (8 + r) : r);
#pragma unroll
    for (int nf = 0; nf < 4; ++nf) {
      const int d = (lane & 15) + 16 * nf;
      outh[((size_t)b * Ss + q) * Ee + h * HDd + d] = (_Float16)(o[nf][r] * lr);
    }
  }
}

// ---------------------------------------------------------------------------
extern "C" void kernel_launch(void* const* d_in, const int* in_sizes, int n_in,
                              void* d_out, int out_size, void* d_ws, size_t ws_size,
                              hipStream_t stream) {
  const float* x  = (const float*)d_in[0];
  const float* Wq = (const float*)d_in[1];
  const float* bq = (const float*)d_in[2];
  const float* Wk = (const float*)d_in[3];
  const float* bk = (const float*)d_in[4];
  const float* Wv = (const float*)d_in[5];
  const float* bv = (const float*)d_in[6];
  const float* Wo = (const float*)d_in[7];
  const float* bo = (const float*)d_in[8];

  char* ws = (char*)d_ws;
  _Float16* xh  = (_Float16*)(ws);                       // 8 MiB
  _Float16* wqh = (_Float16*)(ws + (size_t)(8u  << 20)); // 2 MiB each
  _Float16* wkh = (_Float16*)(ws + (size_t)(10u << 20));
  _Float16* wvh = (_Float16*)(ws + (size_t)(12u << 20));
  _Float16* woh = (_Float16*)(ws + (size_t)(14u << 20));
  _Float16* Qh  = (_Float16*)(ws + (size_t)(16u << 20)); // [B,H,S,HD]
  _Float16* Kh  = (_Float16*)(ws + (size_t)(24u << 20)); // [B,H,S,HD]
  _Float16* Vt  = (_Float16*)(ws + (size_t)(32u << 20)); // [B,H,HD,S]
  _Float16* Ah  = (_Float16*)(ws + (size_t)(40u << 20)); // [B,S,E]

  const int nX = Mtot * Ee, nW = Ee * Ee;
  cast_f32_to_f16<<<nX / 4 / 256, 256, 0, stream>>>(x, xh, nX);
  cast_f32_to_f16<<<nW / 4 / 256, 256, 0, stream>>>(Wq, wqh, nW);
  cast_f32_to_f16<<<nW / 4 / 256, 256, 0, stream>>>(Wk, wkh, nW);
  cast_f32_to_f16<<<nW / 4 / 256, 256, 0, stream>>>(Wv, wvh, nW);
  cast_f32_to_f16<<<nW / 4 / 256, 256, 0, stream>>>(Wo, woh, nW);

  // Fold softmax scale AND log2(e) into Q so attention uses exp2 (v_exp_f32).
  const float qscale = 1.4426950408889634f / 8.0f;  // log2(e)/sqrt(64)

  const int gemmBlocks = (Mtot >> 7) * (Ee >> 7);  // 128x128 tiles
  gemm_wmma_kernel<<<gemmBlocks, 256, 0, stream>>>(xh, wqh, bq, Qh, nullptr,
                                                   Mtot, Ee, Ee, 0, qscale);
  gemm_wmma_kernel<<<gemmBlocks, 256, 0, stream>>>(xh, wkh, bk, Kh, nullptr,
                                                   Mtot, Ee, Ee, 0, 1.0f);
  gemm_wmma_kernel<<<gemmBlocks, 256, 0, stream>>>(xh, wvh, bv, Vt, nullptr,
                                                   Mtot, Ee, Ee, 1, 1.0f);

  flash_attn_kernel<<<512, 256, 0, stream>>>(Qh, Kh, Vt, Ah);

  gemm_wmma_kernel<<<gemmBlocks, 256, 0, stream>>>(Ah, woh, bo, nullptr,
                                                   (float*)d_out, Mtot, Ee, Ee,
                                                   2, 1.0f);
}
